// MambaLayer_58763742544741
// MI455X (gfx1250) — matrix operand order repaired and checked by reading, not compiled, not run on hardware
//
#include <hip/hip_runtime.h>
#include <hip/hip_bf16.h>

// Mamba layer for MI455X (gfx1250), wave32 + WMMA + async LDS staging.
// B=128, L=256, H=256, DI=512, DS=16, DC=4, DTR=16.

#define Bdim 128
#define Ldim 256
#define Hdim 256
#define DIdim 512
#define DSdim 16
#define DCdim 4
#define DTRdim 16
#define ROWS (Bdim * Ldim) // 32768

typedef __bf16 bf16;
typedef __bf16 v16bf __attribute__((ext_vector_type(16)));
typedef __bf16 v8bf  __attribute__((ext_vector_type(8)));
typedef float  v8f   __attribute__((ext_vector_type(8)));
typedef int    v4i   __attribute__((vector_size(16)));

#define AS1 __attribute__((address_space(1)))
#define AS3 __attribute__((address_space(3)))

// ---------------------------------------------------------------------------
// Async global->LDS copy (CDNA5 GLOBAL_LOAD_ASYNC_TO_LDS_B128, ASYNCcnt).
// Builtin signature (from clang diagnostic): (v4i AS1* src, v4i AS3* dst,
// imm offset, imm cpol). Falls back to a synchronous copy if unavailable.
// ---------------------------------------------------------------------------
#if __has_builtin(__builtin_amdgcn_global_load_async_to_lds_b128)
__device__ inline void async_b128(const bf16* src, bf16* dst) {
    __builtin_amdgcn_global_load_async_to_lds_b128(
        (AS1 v4i*)src, (AS3 v4i*)dst, 0, 0);
}
__device__ inline void async_wait0() {
#if __has_builtin(__builtin_amdgcn_s_wait_asynccnt)
    __builtin_amdgcn_s_wait_asynccnt(0);
#else
    asm volatile("s_wait_asynccnt 0x0" ::: "memory");
#endif
}
#else
__device__ inline void async_b128(const bf16* src, bf16* dst) {
    *reinterpret_cast<v8bf*>(dst) = *reinterpret_cast<const v8bf*>(src);
}
__device__ inline void async_wait0() {}
#endif

// ---------------------------------------------------------------------------
// Elementwise f32 -> bf16 convert
// ---------------------------------------------------------------------------
__global__ void f32_to_bf16_kernel(const float* __restrict__ src,
                                   bf16* __restrict__ dst, int n) {
    int i = blockIdx.x * blockDim.x + threadIdx.x;
    if (i < n) dst[i] = (bf16)src[i];
}

// ---------------------------------------------------------------------------
// Transpose + convert: W (K x srcLd f32, cols [colOff, colOff+N)) -> Wt (N x K bf16)
// ---------------------------------------------------------------------------
__global__ void transpose_bf16_kernel(const float* __restrict__ W,
                                      bf16* __restrict__ Wt,
                                      int K, int N, int colOff, int srcLd) {
    int i = blockIdx.x * blockDim.x + threadIdx.x;
    if (i >= N * K) return;
    int n = i / K, k = i % K;
    Wt[i] = (bf16)W[(size_t)k * srcLd + colOff + n];
}

// ---------------------------------------------------------------------------
// W_eff[k][n] = sum_t W_x[k][t] * W_dt[t][n]   (512x512 = 512x16 @ 16x512)
// ---------------------------------------------------------------------------
__global__ void weff_kernel(const float* __restrict__ W_x,
                            const float* __restrict__ W_dt,
                            float* __restrict__ W_eff) {
    int i = blockIdx.x * blockDim.x + threadIdx.x;
    if (i >= DIdim * DIdim) return;
    int k = i / DIdim, n = i % DIdim;
    float acc = 0.f;
#pragma unroll
    for (int t = 0; t < DTRdim; ++t)
        acc += W_x[k * (DTRdim + 2 * DSdim) + t] * W_dt[t * DIdim + n];
    W_eff[i] = acc;
}

// ---------------------------------------------------------------------------
// Fragment loaders (CDNA5 ISA 7.12.2, lane = g*16 + r):
//   A  16x32: lane holds row r, K in [k0+g*8, +8) and [k0+16+g*8, +8)  (2x b128)
//   B  32x16: lane holds col r, K in [k0+g*8, +8) and [k0+16+g*8, +8)  (2x b128)
//   C/D 16x16 f32: elem j -> row g*8+j, col r
// ---------------------------------------------------------------------------
__device__ inline v16bf load_a_frag(const bf16* __restrict__ A, int lda,
                                    int row, int k0, int g) {
    union { v16bf v; v8bf h[2]; } u;
    const bf16* p = A + (size_t)row * lda + k0 + g * 8;
    u.h[0] = *reinterpret_cast<const v8bf*>(p);
    u.h[1] = *reinterpret_cast<const v8bf*>(p + 16);
    return u.v;
}

__device__ inline v16bf load_b_frag(const bf16* __restrict__ Bt, int ldb,
                                    int col, int k0, int g) {
    union { v16bf v; v8bf h[2]; } u;
    const bf16* p = Bt + (size_t)col * ldb + k0 + g * 8;
    u.h[0] = *reinterpret_cast<const v8bf*>(p);
    u.h[1] = *reinterpret_cast<const v8bf*>(p + 16);
    return u.v;
}

// LDS-tile fragment loaders (tile stored row-major, 32 halfs per row)
__device__ inline v16bf lds_a_frag(const bf16* base, int row, int g) {
    union { v16bf v; v8bf h[2]; } u;
    const bf16* p = base + row * 32 + g * 8;
    u.h[0] = *reinterpret_cast<const v8bf*>(p);
    u.h[1] = *reinterpret_cast<const v8bf*>(p + 16);
    return u.v;
}
__device__ inline v16bf lds_b_frag(const bf16* base, int col, int g) {
    union { v16bf v; v8bf h[2]; } u;
    const bf16* p = base + col * 32 + g * 8;
    u.h[0] = *reinterpret_cast<const v8bf*>(p);
    u.h[1] = *reinterpret_cast<const v8bf*>(p + 16);
    return u.v;
}

// ---------------------------------------------------------------------------
// Direct WMMA GEMM (no staging) — used for the narrow N=32 GEMM.
// C[M x N, f32] = A[M x K, bf16 row-major] * Bt[N x K, bf16]^T
// ---------------------------------------------------------------------------
__global__ __launch_bounds__(256)
void wmma_gemm_bf16_kernel(const bf16* __restrict__ A,
                           const bf16* __restrict__ Bt,
                           float* __restrict__ C,
                           int M, int N, int K) {
    const int wave   = blockIdx.x * (blockDim.x >> 5) + (threadIdx.x >> 5);
    const int lane   = threadIdx.x & 31;
    const int tilesN = N >> 5;
    const int tilesM = M >> 5;
    if (wave >= tilesM * tilesN) return;   // uniform per wave: EXEC stays all-ones
    const int m0 = (wave / tilesN) * 32;
    const int n0 = (wave % tilesN) * 32;
    const int g = lane >> 4, r = lane & 15;

    v8f acc00 = {}, acc01 = {}, acc10 = {}, acc11 = {};

    for (int k0 = 0; k0 < K; k0 += 32) {
        v16bf a0 = load_a_frag(A, K, m0 + r,      k0, g);
        v16bf a1 = load_a_frag(A, K, m0 + 16 + r, k0, g);
        v16bf b0 = load_b_frag(Bt, K, n0 + r,      k0, g);
        v16bf b1 = load_b_frag(Bt, K, n0 + 16 + r, k0, g);
        acc00 = __builtin_amdgcn_wmma_f32_16x16x32_bf16(false, a0, false, b0,
                                                        (short)0, acc00, false, false);
        acc01 = __builtin_amdgcn_wmma_f32_16x16x32_bf16(false, a0, false, b1,
                                                        (short)0, acc01, false, false);
        acc10 = __builtin_amdgcn_wmma_f32_16x16x32_bf16(false, a1, false, b0,
                                                        (short)0, acc10, false, false);
        acc11 = __builtin_amdgcn_wmma_f32_16x16x32_bf16(false, a1, false, b1,
                                                        (short)0, acc11, false, false);
    }

#pragma unroll
    for (int j = 0; j < 8; ++j) {
        int r0 = m0 + g * 8 + j;
        int r1 = r0 + 16;
        C[(size_t)r0 * N + n0 + r]      = acc00[j];
        C[(size_t)r0 * N + n0 + 16 + r] = acc01[j];
        C[(size_t)r1 * N + n0 + r]      = acc10[j];
        C[(size_t)r1 * N + n0 + 16 + r] = acc11[j];
    }
}

// ---------------------------------------------------------------------------
// LDS-staged WMMA GEMM with double-buffered async global->LDS copies.
// Block = 256 threads = 8 waves arranged 4(M) x 2(N); block tile 128 x 64.
// Per K-step: stage A 128x32 (8KB) + B 64x32 (4KB); 24KB LDS double-buffered.
// Requires M % 128 == 0, N % 64 == 0, K % 32 == 0 (no partial blocks -> EXEC
// stays all-ones for every WMMA).
// ---------------------------------------------------------------------------
__global__ __launch_bounds__(256)
void wmma_gemm_lds_kernel(const bf16* __restrict__ A,
                          const bf16* __restrict__ Bt,
                          float* __restrict__ C,
                          int M, int N, int K) {
    __shared__ bf16 Atile[2][128 * 32];
    __shared__ bf16 Btile[2][64 * 32];

    const int tilesN = N >> 6;
    const int m0blk = (blockIdx.x / tilesN) * 128;
    const int n0blk = (blockIdx.x % tilesN) * 64;

    const int t    = threadIdx.x;
    const int wave = t >> 5;
    const int lane = t & 31;
    const int wm   = wave >> 1;      // 0..3
    const int wn   = wave & 1;       // 0..1
    const int g = lane >> 4, r = lane & 15;

    auto issue_tiles = [&](int buf, int k0) {
        // A tile: 8KB = 512 x 16B chunks; 2 per thread
#pragma unroll
        for (int j = 0; j < 2; ++j) {
            int e   = t + j * 256;
            int row = e >> 2;        // 0..127
            int ch  = e & 3;         // 16B chunk within 64B row
            async_b128(A + (size_t)(m0blk + row) * K + k0 + ch * 8,
                       &Atile[buf][row * 32 + ch * 8]);
        }
        // B tile: 4KB = 256 x 16B chunks; 1 per thread
        {
            int row = t >> 2;        // 0..63
            int ch  = t & 3;
            async_b128(Bt + (size_t)(n0blk + row) * K + k0 + ch * 8,
                       &Btile[buf][row * 32 + ch * 8]);
        }
    };

    v8f acc00 = {}, acc01 = {}, acc10 = {}, acc11 = {};

    issue_tiles(0, 0);
    const int KT = K >> 5;
    for (int kt = 0; kt < KT; ++kt) {
        const int cur = kt & 1;
        async_wait0();               // my share of buf `cur` has landed in LDS
        __syncthreads();             // everyone's share landed; prior reads done
        if (kt + 1 < KT) issue_tiles(1 - cur, (kt + 1) << 5);

        v16bf a0 = lds_a_frag(Atile[cur], wm * 32 + r,      g);
        v16bf a1 = lds_a_frag(Atile[cur], wm * 32 + 16 + r, g);
        v16bf b0 = lds_b_frag(Btile[cur], wn * 32 + r,      g);
        v16bf b1 = lds_b_frag(Btile[cur], wn * 32 + 16 + r, g);
        acc00 = __builtin_amdgcn_wmma_f32_16x16x32_bf16(false, a0, false, b0,
                                                        (short)0, acc00, false, false);
        acc01 = __builtin_amdgcn_wmma_f32_16x16x32_bf16(false, a0, false, b1,
                                                        (short)0, acc01, false, false);
        acc10 = __builtin_amdgcn_wmma_f32_16x16x32_bf16(false, a1, false, b0,
                                                        (short)0, acc10, false, false);
        acc11 = __builtin_amdgcn_wmma_f32_16x16x32_bf16(false, a1, false, b1,
                                                        (short)0, acc11, false, false);
    }

    const int m0 = m0blk + wm * 32;
    const int n0 = n0blk + wn * 32;
#pragma unroll
    for (int j = 0; j < 8; ++j) {
        int r0 = m0 + g * 8 + j;
        int r1 = r0 + 16;
        C[(size_t)r0 * N + n0 + r]      = acc00[j];
        C[(size_t)r0 * N + n0 + 16 + r] = acc01[j];
        C[(size_t)r1 * N + n0 + r]      = acc10[j];
        C[(size_t)r1 * N + n0 + 16 + r] = acc11[j];
    }
}

// ---------------------------------------------------------------------------
// Depthwise causal conv (DC=4) over L + bias + SiLU; emits f32 and bf16.
// x is the first DI columns of XZ (row stride 2*DI).
// ---------------------------------------------------------------------------
__global__ void conv_silu_kernel(const float* __restrict__ XZ,
                                 const float* __restrict__ conv_w, // (DC,1,DI)
                                 const float* __restrict__ conv_b,
                                 float* __restrict__ xc,
                                 bf16* __restrict__ xcb) {
    int i = blockIdx.x * blockDim.x + threadIdx.x;
    if (i >= ROWS * DIdim) return;
    int d = i % DIdim;
    int l = (i / DIdim) % Ldim;
    int b = i / (DIdim * Ldim);
    float acc = conv_b[d];
#pragma unroll
    for (int k = 0; k < DCdim; ++k) {
        int l2 = l + k - (DCdim - 1);
        if (l2 >= 0)
            acc += XZ[((size_t)(b * Ldim + l2)) * (2 * DIdim) + d] * conv_w[k * DIdim + d];
    }
    float v = acc / (1.f + expf(-acc));   // SiLU
    xc[i]  = v;
    xcb[i] = (bf16)v;
}

// ---------------------------------------------------------------------------
// dt = softplus(dt_pre + b_dt), numerically stable, in place.
// ---------------------------------------------------------------------------
__global__ void softplus_kernel(float* __restrict__ dt,
                                const float* __restrict__ b_dt, int n) {
    int i = blockIdx.x * blockDim.x + threadIdx.x;
    if (i >= n) return;
    float v = dt[i] + b_dt[i % DIdim];
    dt[i] = fmaxf(v, 0.f) + log1pf(expf(-fabsf(v)));
}

// ---------------------------------------------------------------------------
// Selective scan. Block = 256 threads = 256 channels of one batch (2 blocks/b).
// h[DS] in registers; B_t/C_t broadcast through LDS each step.
// ---------------------------------------------------------------------------
__global__ __launch_bounds__(256)
void scan_kernel(const float* __restrict__ xc,    // (ROWS, DI)
                 const float* __restrict__ dt,    // (ROWS, DI)
                 const float* __restrict__ BC,    // (ROWS, 32) : B then C
                 const float* __restrict__ A_log, // (DI, DS)
                 float* __restrict__ y) {         // (ROWS, DI)
    const int b  = blockIdx.x >> 1;
    const int d  = ((blockIdx.x & 1) << 8) + threadIdx.x;
    __shared__ float sBC[2 * DSdim];

    float a[DSdim], h[DSdim];
#pragma unroll
    for (int s = 0; s < DSdim; ++s) {
        a[s] = -expf(A_log[d * DSdim + s]);
        h[s] = 0.f;
    }

    for (int l = 0; l < Ldim; ++l) {
        const size_t row = (size_t)b * Ldim + l;
        if (threadIdx.x < 2 * DSdim) sBC[threadIdx.x] = BC[row * (2 * DSdim) + threadIdx.x];
        __syncthreads();
        const float x_t  = xc[row * DIdim + d];
        const float dt_t = dt[row * DIdim + d];
        const float dtx  = dt_t * x_t;
        float yt = 0.f;
#pragma unroll
        for (int s = 0; s < DSdim; ++s) {
            float dA = expf(dt_t * a[s]);
            h[s] = h[s] * dA + dtx * sBC[s];
            yt += h[s] * sBC[DSdim + s];
        }
        y[row * DIdim + d] = yt;
        __syncthreads();
    }
}

// ---------------------------------------------------------------------------
// y2 = (y + xc*D) * silu(z); emitted as bf16 for the output GEMM.
// ---------------------------------------------------------------------------
__global__ void gate_kernel(const float* __restrict__ y,
                            const float* __restrict__ xc,
                            const float* __restrict__ Dvec,
                            const float* __restrict__ XZ,
                            bf16* __restrict__ yb) {
    int i = blockIdx.x * blockDim.x + threadIdx.x;
    if (i >= ROWS * DIdim) return;
    int d = i % DIdim;
    size_t row = (size_t)(i / DIdim);
    float z = XZ[row * (2 * DIdim) + DIdim + d];
    float sz = z / (1.f + expf(-z));
    yb[i] = (bf16)((y[i] + xc[i] * Dvec[d]) * sz);
}

// ---------------------------------------------------------------------------
// Residual + LayerNorm: one block (256 threads) per row of H=256.
// ---------------------------------------------------------------------------
__global__ __launch_bounds__(256)
void layernorm_kernel(const float* __restrict__ out_pre,
                      const float* __restrict__ inp,
                      const float* __restrict__ g,
                      const float* __restrict__ bvec,
                      float* __restrict__ out) {
    const size_t row = blockIdx.x;
    const int t = threadIdx.x;
    __shared__ float ssum[Hdim];
    __shared__ float ssq[Hdim];
    float v = out_pre[row * Hdim + t] + inp[row * Hdim + t];
    ssum[t] = v;
    ssq[t]  = v * v;
    __syncthreads();
    for (int off = Hdim >> 1; off > 0; off >>= 1) {
        if (t < off) { ssum[t] += ssum[t + off]; ssq[t] += ssq[t + off]; }
        __syncthreads();
    }
    float mean = ssum[0] * (1.f / Hdim);
    float var  = ssq[0] * (1.f / Hdim) - mean * mean;
    float rstd = rsqrtf(var + 1e-12f);
    out[row * Hdim + t] = (v - mean) * rstd * g[t] + bvec[t];
}

// ---------------------------------------------------------------------------
// Host-side launch
// ---------------------------------------------------------------------------
extern "C" void kernel_launch(void* const* d_in, const int* in_sizes, int n_in,
                              void* d_out, int out_size, void* d_ws, size_t ws_size,
                              hipStream_t stream) {
    const float* inp    = (const float*)d_in[0];   // (B,L,H)
    const float* W_in   = (const float*)d_in[1];   // (H, 2*DI)
    const float* conv_w = (const float*)d_in[2];   // (DC,1,DI)
    const float* conv_b = (const float*)d_in[3];   // (DI)
    const float* W_x    = (const float*)d_in[4];   // (DI, DTR+2*DS)
    const float* W_dt   = (const float*)d_in[5];   // (DTR, DI)
    const float* b_dt   = (const float*)d_in[6];   // (DI)
    const float* A_log  = (const float*)d_in[7];   // (DI, DS)
    const float* Dvec   = (const float*)d_in[8];   // (DI)
    const float* W_out  = (const float*)d_in[9];   // (DI, H)
    const float* ln_g   = (const float*)d_in[10];  // (H)
    const float* ln_b   = (const float*)d_in[11];  // (H)
    float* out = (float*)d_out;

    // Linear carve of workspace (256-B aligned); ~440 MB total.
    char* base = (char*)d_ws;
    size_t off = 0;
    auto carve = [&](size_t bytes) -> char* {
        char* p = base + off;
        off += (bytes + 255) & ~(size_t)255;
        return p;
    };
    bf16*  Xb      = (bf16*) carve((size_t)ROWS * Hdim * 2);          // 16 MB
    bf16*  Wt_in   = (bf16*) carve((size_t)(2 * DIdim) * Hdim * 2);   // 0.5 MB
    float* XZ      = (float*)carve((size_t)ROWS * 2 * DIdim * 4);     // 128 MB
    float* xc      = (float*)carve((size_t)ROWS * DIdim * 4);         // 64 MB
    bf16*  xcb     = (bf16*) carve((size_t)ROWS * DIdim * 2);         // 32 MB
    bf16*  Wt_bc   = (bf16*) carve((size_t)(2 * DSdim) * DIdim * 2);
    float* W_eff   = (float*)carve((size_t)DIdim * DIdim * 4);        // 1 MB
    bf16*  Wt_eff  = (bf16*) carve((size_t)DIdim * DIdim * 2);
    bf16*  Wt_out  = (bf16*) carve((size_t)Hdim * DIdim * 2);
    float* BC      = (float*)carve((size_t)ROWS * 2 * DSdim * 4);     // 4 MB
    float* dtbuf   = (float*)carve((size_t)ROWS * DIdim * 4);         // 64 MB
    float* ybuf    = (float*)carve((size_t)ROWS * DIdim * 4);         // 64 MB
    bf16*  yb      = (bf16*) carve((size_t)ROWS * DIdim * 2);         // 32 MB
    float* out_pre = (float*)carve((size_t)ROWS * Hdim * 4);          // 32 MB

    const int TPB = 256;
    auto cdiv = [](long a, long b) { return (int)((a + b - 1) / b); };
    auto lds_gemm_blocks = [&](int M, int N) { return (M / 128) * (N / 64); };

    // 1) X -> bf16
    f32_to_bf16_kernel<<<cdiv((long)ROWS * Hdim, TPB), TPB, 0, stream>>>(inp, Xb, ROWS * Hdim);
    // 2) W_in^T -> bf16 (2DI x H)
    transpose_bf16_kernel<<<cdiv((long)2 * DIdim * Hdim, TPB), TPB, 0, stream>>>(
        W_in, Wt_in, Hdim, 2 * DIdim, 0, 2 * DIdim);
    // 3) XZ = X @ W_in   (32768 x 1024, K=256) — LDS-staged WMMA
    wmma_gemm_lds_kernel<<<lds_gemm_blocks(ROWS, 2 * DIdim), TPB, 0, stream>>>(
        Xb, Wt_in, XZ, ROWS, 2 * DIdim, Hdim);
    // 4) causal depthwise conv + SiLU (emits f32 + bf16)
    conv_silu_kernel<<<cdiv((long)ROWS * DIdim, TPB), TPB, 0, stream>>>(
        XZ, conv_w, conv_b, xc, xcb);
    // 5) W_x columns [16,48)^T -> bf16 (32 x 512)
    transpose_bf16_kernel<<<cdiv((long)2 * DSdim * DIdim, TPB), TPB, 0, stream>>>(
        W_x, Wt_bc, DIdim, 2 * DSdim, DTRdim, DTRdim + 2 * DSdim);
    // 6) BC = xc @ W_x[:,16:48]   (32768 x 32, K=512) — direct WMMA (narrow N)
    wmma_gemm_bf16_kernel<<<cdiv((long)(ROWS / 32) * (2 * DSdim / 32), 8), TPB, 0, stream>>>(
        xcb, Wt_bc, BC, ROWS, 2 * DSdim, DIdim);
    // 7) W_eff = W_x[:, :16] @ W_dt  (512 x 512)
    weff_kernel<<<cdiv((long)DIdim * DIdim, TPB), TPB, 0, stream>>>(W_x, W_dt, W_eff);
    // 8) W_eff^T -> bf16
    transpose_bf16_kernel<<<cdiv((long)DIdim * DIdim, TPB), TPB, 0, stream>>>(
        W_eff, Wt_eff, DIdim, DIdim, 0, DIdim);
    // 9) dt_pre = xc @ W_eff   (32768 x 512, K=512) — LDS-staged WMMA
    wmma_gemm_lds_kernel<<<lds_gemm_blocks(ROWS, DIdim), TPB, 0, stream>>>(
        xcb, Wt_eff, dtbuf, ROWS, DIdim, DIdim);
    // 10) dt = softplus(dt_pre + b_dt)
    softplus_kernel<<<cdiv((long)ROWS * DIdim, TPB), TPB, 0, stream>>>(dtbuf, b_dt, ROWS * DIdim);
    // 11) selective scan
    scan_kernel<<<Bdim * (DIdim / 256), 256, 0, stream>>>(xc, dtbuf, BC, A_log, ybuf);
    // 12) gate: yb = bf16((y + xc*D) * silu(z))
    gate_kernel<<<cdiv((long)ROWS * DIdim, TPB), TPB, 0, stream>>>(ybuf, xc, Dvec, XZ, yb);
    // 13) W_out^T -> bf16 (256 x 512)
    transpose_bf16_kernel<<<cdiv((long)Hdim * DIdim, TPB), TPB, 0, stream>>>(
        W_out, Wt_out, DIdim, Hdim, 0, Hdim);
    // 14) out_pre = y2 @ W_out   (32768 x 256, K=512) — LDS-staged WMMA
    wmma_gemm_lds_kernel<<<lds_gemm_blocks(ROWS, Hdim), TPB, 0, stream>>>(
        yb, Wt_out, out_pre, ROWS, Hdim, DIdim);
    // 15) residual + LayerNorm -> d_out
    layernorm_kernel<<<ROWS, Hdim, 0, stream>>>(out_pre, inp, ln_g, ln_b, out);
}
